// SlidingWindowAttention_3736621547664
// MI455X (gfx1250) — matrix operand order repaired and checked by reading, not compiled, not run on hardware
//
#include <hip/hip_runtime.h>
#include <hip/hip_bf16.h>

typedef __attribute__((ext_vector_type(16))) _Float16 v16h;
typedef __attribute__((ext_vector_type(8)))  _Float16 v8h;
typedef __attribute__((ext_vector_type(8)))  float    v8f;
typedef unsigned int u32x4 __attribute__((ext_vector_type(4)));
typedef int          i32x8 __attribute__((ext_vector_type(8)));
typedef int          i32x4 __attribute__((ext_vector_type(4)));

#define T_SEQ 2048
#define C_DIM 1024
#define N_HEADS 16
#define D_HEAD 64
#define WINDOW 256
#define BH 32           // B * H
#define M_ROWS 4096     // B * T

// ---------------------------------------------------------------------------
// WMMA helpers (gfx1250, wave32). Layouts per CDNA5 ISA 7.12.2.
// ---------------------------------------------------------------------------
__device__ __forceinline__ v8f wmma_f16(v16h a, v16h b, v8f c) {
    return __builtin_amdgcn_wmma_f32_16x16x32_f16(false, a, false, b, (short)0, c, false, false);
}

// A fragment: 16x32 f16 of a row-major [.., ld] matrix.
__device__ __forceinline__ v16h load_frag_a(const _Float16* base, int ld, int row0, int k0) {
    int lane = threadIdx.x & 31;
    int m = lane & 15, half = lane >> 4;
    const _Float16* p = base + (size_t)(row0 + m) * ld + k0 + half * 16;
    v8h lo = *(const v8h*)(p);
    v8h hi = *(const v8h*)(p + 8);
    return __builtin_shufflevector(lo, hi, 0,1,2,3,4,5,6,7,8,9,10,11,12,13,14,15);
}

// B fragment: 32x16 (KxN) from an N-major matrix (row n contiguous in K).
__device__ __forceinline__ v16h load_frag_b(const _Float16* base, int ld, int col0, int k0) {
    int lane = threadIdx.x & 31;
    int n = lane & 15, half = lane >> 4;
    return *(const v16h*)(base + (size_t)(col0 + n) * ld + k0 + half * 16);
}

__device__ __forceinline__ float rmax16(float v) {
    v = fmaxf(v, __shfl_xor(v, 1, 32));
    v = fmaxf(v, __shfl_xor(v, 2, 32));
    v = fmaxf(v, __shfl_xor(v, 4, 32));
    v = fmaxf(v, __shfl_xor(v, 8, 32));
    return v;
}
__device__ __forceinline__ float rsum16(float v) {
    v += __shfl_xor(v, 1, 32);
    v += __shfl_xor(v, 2, 32);
    v += __shfl_xor(v, 4, 32);
    v += __shfl_xor(v, 8, 32);
    return v;
}

// ---------------------------------------------------------------------------
// TDM: 2D tile load Global -> LDS. Descriptor per CDNA5 ISA 08_async_tensor §8.
// group0: count=1 | lds_addr | global_addr[56:0] | type=2 (bits 127:126)
// group1: data_size=1 (2B) | tensor_dim0/1 | tile_dim0/1 | tensor_dim0_stride
// groups 2/3 + extra dwords: zero (2-D tensor). Tracked with TENSORcnt.
// The LDS destination pointer is captured by an empty asm with a memory
// clobber so the compiler knows the LDS array is written (otherwise loads
// from it fold to undef, since the TDM write is invisible to alias analysis).
// ---------------------------------------------------------------------------
__device__ __forceinline__ void tdm_load_2d(_Float16* lds_ptr, const void* gptr,
                                            unsigned tile_d0, unsigned tile_d1,
                                            unsigned tensor_d0, unsigned tensor_d1,
                                            unsigned long long stride0) {
    // Generic LDS addresses are {SHARED_BASE[63:32], offset[31:0]} -> low 32
    // bits are the byte offset within the workgroup's LDS allocation.
    unsigned lds_byte_off = (unsigned)(uintptr_t)lds_ptr;
    unsigned long long ga = (unsigned long long)(uintptr_t)gptr;
    u32x4 g0;
    g0.x = 1u;                                                   // count=1, user mode
    g0.y = lds_byte_off;                                         // LDS byte address
    g0.z = (unsigned)ga;                                         // global addr [31:0]
    g0.w = (unsigned)((ga >> 32) & 0x01FFFFFFu) | (2u << 30);    // [56:32] | type=2
    i32x8 g1;
    g1[0] = (int)(1u << 16);                                     // data_size=1 -> 2 bytes
    g1[1] = (int)((tensor_d0 & 0xFFFFu) << 16);                  // tensor_dim0[15:0]
    g1[2] = (int)((tensor_d0 >> 16) | ((tensor_d1 & 0xFFFFu) << 16));
    g1[3] = (int)((tensor_d1 >> 16) | (tile_d0 << 16));          // | tile_dim0
    g1[4] = (int)(tile_d1 & 0xFFFFu);                            // tile_dim1, tile_dim2=0
    g1[5] = (int)(unsigned)(stride0 & 0xFFFFFFFFull);            // dim0 stride lo
    g1[6] = (int)(unsigned)((stride0 >> 32) & 0xFFFFull);        // dim0 stride hi
    g1[7] = 0;
    i32x4 z4 = {0, 0, 0, 0};
    i32x8 z8 = {0, 0, 0, 0, 0, 0, 0, 0};
    __builtin_amdgcn_tensor_load_to_lds(g0, g1, z4, z4, z8, 0);
    // Escape the LDS pointer: TDM writes this memory behind the compiler's back.
    asm volatile("" : : "v"(lds_ptr) : "memory");
}

// ---------------------------------------------------------------------------
// 1) RMSNorm: one block per row, output f16
// ---------------------------------------------------------------------------
__global__ void rmsnorm_kernel(const float* __restrict__ x, _Float16* __restrict__ xn) {
    int row = blockIdx.x;
    const float* xr = x + (size_t)row * C_DIM;
    float ss = 0.f;
    for (int i = threadIdx.x; i < C_DIM; i += 256) { float v = xr[i]; ss += v * v; }
    for (int off = 16; off; off >>= 1) ss += __shfl_xor(ss, off, 32);
    __shared__ float wsum[8];
    if ((threadIdx.x & 31) == 0) wsum[threadIdx.x >> 5] = ss;
    __syncthreads();
    float tot = 0.f;
    #pragma unroll
    for (int w = 0; w < 8; w++) tot += wsum[w];
    float scale = rsqrtf(tot * (1.0f / C_DIM) + 1e-6f);
    _Float16* dr = xn + (size_t)row * C_DIM;
    for (int i = threadIdx.x; i < C_DIM; i += 256) dr[i] = (_Float16)(xr[i] * scale);
}

// ---------------------------------------------------------------------------
// 2) f32 -> f16 conversion
// ---------------------------------------------------------------------------
__global__ void cvt_f32_f16_kernel(const float* __restrict__ s, _Float16* __restrict__ d, int n) {
    int i = blockIdx.x * 256 + threadIdx.x;
    if (i < n) d[i] = (_Float16)s[i];
}

// ---------------------------------------------------------------------------
// GEMM core with TDM double-buffered LDS staging.
// Block = 8 waves, block tile 128(M) x 64(N), K chunk = 64.
// LDS: A 2x[128x64] f16 (2x16KB), B 2x[64x64] f16 (2x8KB) = 48KB.
// Wave tile 32x32: 4 WMMA per 32-K sub-step, 8 per chunk, 128 total.
// ---------------------------------------------------------------------------
#define KC 64
#define SA_HALVES (128 * KC)      // 8192
#define SB_HALVES (64 * KC)       // 4096

// ---------------------------------------------------------------------------
// 3) QKV GEMM: [4096,1024] x w_qkv^T -> scatter into q,k [BH,T,D], vT [BH,D,T]
// ---------------------------------------------------------------------------
__global__ void qkv_gemm_kernel(const _Float16* __restrict__ xn, const _Float16* __restrict__ w,
                                _Float16* __restrict__ q, _Float16* __restrict__ k,
                                _Float16* __restrict__ vT) {
    __shared__ _Float16 smem[2 * SA_HALVES + 2 * SB_HALVES];
    int wid = threadIdx.x >> 5, lane = threadIdx.x & 31;
    int m = lane & 15, half = lane >> 4;
    int row_block = blockIdx.x * 128;
    int col_block = blockIdx.y * 64;
    int mrow = (wid >> 1) * 32;
    int ncol = (wid & 1) * 32;
    v8f acc[2][2] = {};

    // prologue: stage chunk 0
    if (wid == 0) {
        tdm_load_2d(smem,                  xn + (size_t)row_block * C_DIM, KC, 128, C_DIM, M_ROWS, C_DIM);
        tdm_load_2d(smem + 2 * SA_HALVES,  w  + (size_t)col_block * C_DIM, KC, 64,  C_DIM, 3 * C_DIM, C_DIM);
        __builtin_amdgcn_s_wait_tensorcnt(0);
    }
    __syncthreads();

    for (int kc = 0; kc < C_DIM; kc += KC) {
        int cur = (kc / KC) & 1, nxt = cur ^ 1;
        if (wid == 0 && kc + KC < C_DIM) {
            tdm_load_2d(smem + nxt * SA_HALVES,
                        xn + (size_t)row_block * C_DIM + kc + KC, KC, 128, C_DIM, M_ROWS, C_DIM);
            tdm_load_2d(smem + 2 * SA_HALVES + nxt * SB_HALVES,
                        w + (size_t)col_block * C_DIM + kc + KC, KC, 64, C_DIM, 3 * C_DIM, C_DIM);
        }
        const _Float16* la = smem + cur * SA_HALVES;
        const _Float16* lb = smem + 2 * SA_HALVES + cur * SB_HALVES;
        #pragma unroll
        for (int ks = 0; ks < KC; ks += 32) {
            v16h a0 = load_frag_a(la, KC, mrow,      ks);
            v16h a1 = load_frag_a(la, KC, mrow + 16, ks);
            v16h b0 = load_frag_b(lb, KC, ncol,      ks);
            v16h b1 = load_frag_b(lb, KC, ncol + 16, ks);
            acc[0][0] = wmma_f16(a0, b0, acc[0][0]);
            acc[0][1] = wmma_f16(a0, b1, acc[0][1]);
            acc[1][0] = wmma_f16(a1, b0, acc[1][0]);
            acc[1][1] = wmma_f16(a1, b1, acc[1][1]);
        }
        if (wid == 0 && kc + KC < C_DIM) __builtin_amdgcn_s_wait_tensorcnt(0);
        __syncthreads();
    }

    #pragma unroll
    for (int i = 0; i < 2; i++)
    #pragma unroll
    for (int j = 0; j < 2; j++)
    #pragma unroll
    for (int r = 0; r < 8; r++) {
        int row = row_block + mrow + i * 16 + r + 8 * half;   // token index
        int col = col_block + ncol + j * 16 + m;              // 0..3071
        _Float16 val = (_Float16)acc[i][j][r];
        int which = col >> 10, win = col & 1023;
        int hh = win >> 6, d = win & 63;
        int bb = row >> 11, t = row & (T_SEQ - 1);
        size_t bh = (size_t)(bb * N_HEADS + hh);
        if (which == 0)      q[(bh * T_SEQ + t) * D_HEAD + d] = val;
        else if (which == 1) k[(bh * T_SEQ + t) * D_HEAD + d] = val;
        else                 vT[(bh * D_HEAD + d) * T_SEQ + t] = val;
    }
}

// ---------------------------------------------------------------------------
// 4) RoPE in place on q,k [BH,T,D], rotate pairs (i, i+32)
// ---------------------------------------------------------------------------
__global__ void rope_kernel(_Float16* __restrict__ q, _Float16* __restrict__ k) {
    int idx = blockIdx.x * 256 + threadIdx.x;   // BH * T * 32
    int i  = idx & 31;
    int t  = (idx >> 5) & (T_SEQ - 1);
    int bh = idx >> 16;
    float inv = powf(10000.f, -(float)i / 32.f);
    float ang = (float)t * inv;
    float c = cosf(ang), s = sinf(ang);
    size_t base = ((size_t)bh * T_SEQ + t) * D_HEAD;
    float q1 = q[base + i], q2 = q[base + i + 32];
    q[base + i]      = (_Float16)(q1 * c - q2 * s);
    q[base + i + 32] = (_Float16)(q2 * c + q1 * s);
    float k1 = k[base + i], k2 = k[base + i + 32];
    k[base + i]      = (_Float16)(k1 * c - k2 * s);
    k[base + i + 32] = (_Float16)(k2 * c + k1 * s);
}

// ---------------------------------------------------------------------------
// 5) Flash sliding-window attention
// ---------------------------------------------------------------------------
__global__ void attn_kernel(const _Float16* __restrict__ qf, const _Float16* __restrict__ kf,
                            const _Float16* __restrict__ vT, _Float16* __restrict__ oatt) {
    __shared__ _Float16 lds_p[8 * 16 * 32];
    int wid = threadIdx.x >> 5, lane = threadIdx.x & 31;
    int m = lane & 15, half = lane >> 4;
    int bh = blockIdx.x;
    int b = bh >> 4, h = bh & 15;
    int q0 = blockIdx.y * 128 + wid * 16;
    const _Float16* qb = qf + (size_t)bh * T_SEQ * D_HEAD;
    const _Float16* kb = kf + (size_t)bh * T_SEQ * D_HEAD;
    const _Float16* vb = vT + (size_t)bh * D_HEAD * T_SEQ;
    _Float16* myp = lds_p + wid * 512;

    v16h qa0 = load_frag_a(qb, D_HEAD, q0, 0);
    v16h qa1 = load_frag_a(qb, D_HEAD, q0, 32);

    float mrow[8], lrow[8];
    v8f o[4] = {};
    #pragma unroll
    for (int r = 0; r < 8; r++) { mrow[r] = -3.0e38f; lrow[r] = 0.f; }

    for (int ci = 0; ci < 9; ci++) {
        int c = q0 - WINDOW + ci * 32;
        if (c + 32 <= 0) continue;
        int key0 = c + m,       kc0 = min(max(key0, 0), T_SEQ - 1);
        int key1 = c + 16 + m,  kc1 = min(max(key1, 0), T_SEQ - 1);
        const _Float16* p0 = kb + (size_t)kc0 * D_HEAD + half * 16;
        const _Float16* p1 = kb + (size_t)kc1 * D_HEAD + half * 16;
        v16h kb00 = *(const v16h*)(p0);
        v16h kb01 = *(const v16h*)(p0 + 32);
        v16h kb10 = *(const v16h*)(p1);
        v16h kb11 = *(const v16h*)(p1 + 32);
        v8f s0 = {}, s1 = {};
        s0 = wmma_f16(qa0, kb00, s0); s0 = wmma_f16(qa1, kb01, s0);
        s1 = wmma_f16(qa0, kb10, s1); s1 = wmma_f16(qa1, kb11, s1);

        #pragma unroll
        for (int r = 0; r < 8; r++) {
            int qrow = q0 + r + 8 * half;
            int k0i = c + m, k1i = c + 16 + m;
            float v0 = s0[r] * 0.125f, v1 = s1[r] * 0.125f;
            bool ok0 = (k0i >= 0) && (k0i <= qrow) && (qrow - k0i <= WINDOW);
            bool ok1 = (k1i >= 0) && (k1i <= qrow) && (qrow - k1i <= WINDOW);
            v0 = ok0 ? v0 : -1.0e9f;
            v1 = ok1 ? v1 : -1.0e9f;
            float cm = rmax16(fmaxf(v0, v1));
            float nm = fmaxf(mrow[r], cm);
            float alpha = __expf(mrow[r] - nm);
            mrow[r] = nm;
            float pp0 = __expf(v0 - nm), pp1 = __expf(v1 - nm);
            lrow[r] = lrow[r] * alpha + rsum16(pp0 + pp1);
            myp[(r + 8 * half) * 32 + m]      = (_Float16)pp0;
            myp[(r + 8 * half) * 32 + 16 + m] = (_Float16)pp1;
            #pragma unroll
            for (int dt = 0; dt < 4; dt++) o[dt][r] = o[dt][r] * alpha;
        }
        asm volatile("s_wait_dscnt 0" ::: "memory");

        v8h plo = *(const v8h*)(myp + m * 32 + half * 16);
        v8h phi = *(const v8h*)(myp + m * 32 + half * 16 + 8);
        v16h pa = __builtin_shufflevector(plo, phi, 0,1,2,3,4,5,6,7,8,9,10,11,12,13,14,15);

        int ck = c + half * 16;
        #pragma unroll
        for (int dt = 0; dt < 4; dt++) {
            int d = dt * 16 + m;
            const _Float16* vrow = vb + (size_t)d * T_SEQ;
            v16h vfrag = {};
            if (ck >= 0 && ck + 16 <= T_SEQ) {
                vfrag = *(const v16h*)(vrow + ck);
            } else {
                #pragma unroll
                for (int j = 0; j < 16; j++) {
                    int kk2 = min(max(ck + j, 0), T_SEQ - 1);
                    vfrag[j] = vrow[kk2];
                }
            }
            o[dt] = wmma_f16(pa, vfrag, o[dt]);
        }
    }

    #pragma unroll
    for (int dt = 0; dt < 4; dt++)
    #pragma unroll
    for (int r = 0; r < 8; r++) {
        int qrow = q0 + r + 8 * half;
        float val = o[dt][r] / lrow[r];
        oatt[((size_t)b * T_SEQ + qrow) * C_DIM + h * D_HEAD + dt * 16 + m] = (_Float16)val;
    }
}

// ---------------------------------------------------------------------------
// 6) Output GEMM: attn_out [4096,1024] x w_out^T -> f32 d_out (TDM-staged)
// ---------------------------------------------------------------------------
__global__ void out_gemm_kernel(const _Float16* __restrict__ ao, const _Float16* __restrict__ w,
                                float* __restrict__ out) {
    __shared__ _Float16 smem[2 * SA_HALVES + 2 * SB_HALVES];
    int wid = threadIdx.x >> 5, lane = threadIdx.x & 31;
    int m = lane & 15, half = lane >> 4;
    int row_block = blockIdx.x * 128;
    int col_block = blockIdx.y * 64;
    int mrow = (wid >> 1) * 32;
    int ncol = (wid & 1) * 32;
    v8f acc[2][2] = {};

    if (wid == 0) {
        tdm_load_2d(smem,                 ao + (size_t)row_block * C_DIM, KC, 128, C_DIM, M_ROWS, C_DIM);
        tdm_load_2d(smem + 2 * SA_HALVES, w  + (size_t)col_block * C_DIM, KC, 64,  C_DIM, C_DIM, C_DIM);
        __builtin_amdgcn_s_wait_tensorcnt(0);
    }
    __syncthreads();

    for (int kc = 0; kc < C_DIM; kc += KC) {
        int cur = (kc / KC) & 1, nxt = cur ^ 1;
        if (wid == 0 && kc + KC < C_DIM) {
            tdm_load_2d(smem + nxt * SA_HALVES,
                        ao + (size_t)row_block * C_DIM + kc + KC, KC, 128, C_DIM, M_ROWS, C_DIM);
            tdm_load_2d(smem + 2 * SA_HALVES + nxt * SB_HALVES,
                        w + (size_t)col_block * C_DIM + kc + KC, KC, 64, C_DIM, C_DIM, C_DIM);
        }
        const _Float16* la = smem + cur * SA_HALVES;
        const _Float16* lb = smem + 2 * SA_HALVES + cur * SB_HALVES;
        #pragma unroll
        for (int ks = 0; ks < KC; ks += 32) {
            v16h a0 = load_frag_a(la, KC, mrow,      ks);
            v16h a1 = load_frag_a(la, KC, mrow + 16, ks);
            v16h b0 = load_frag_b(lb, KC, ncol,      ks);
            v16h b1 = load_frag_b(lb, KC, ncol + 16, ks);
            acc[0][0] = wmma_f16(a0, b0, acc[0][0]);
            acc[0][1] = wmma_f16(a0, b1, acc[0][1]);
            acc[1][0] = wmma_f16(a1, b0, acc[1][0]);
            acc[1][1] = wmma_f16(a1, b1, acc[1][1]);
        }
        if (wid == 0 && kc + KC < C_DIM) __builtin_amdgcn_s_wait_tensorcnt(0);
        __syncthreads();
    }

    #pragma unroll
    for (int i = 0; i < 2; i++)
    #pragma unroll
    for (int j = 0; j < 2; j++)
    #pragma unroll
    for (int r = 0; r < 8; r++) {
        int row = row_block + mrow + i * 16 + r + 8 * half;
        int col = col_block + ncol + j * 16 + m;
        out[(size_t)row * C_DIM + col] = acc[i][j][r];
    }
}

// ---------------------------------------------------------------------------
// Workspace layout (all f16): xn@0, wqkv16@8MiB, wout16@14MiB, q@16MiB,
// k@24MiB, vT@32MiB, attn_out@40MiB  (48 MiB total)
// ---------------------------------------------------------------------------
extern "C" void kernel_launch(void* const* d_in, const int* in_sizes, int n_in,
                              void* d_out, int out_size, void* d_ws, size_t ws_size,
                              hipStream_t stream) {
    const float* x     = (const float*)d_in[0];
    const float* w_qkv = (const float*)d_in[1];
    const float* w_out = (const float*)d_in[2];
    float* out = (float*)d_out;

    char* ws = (char*)d_ws;
    _Float16* xn     = (_Float16*)(ws);
    _Float16* wqkv16 = (_Float16*)(ws + (8ull  << 20));
    _Float16* wout16 = (_Float16*)(ws + (14ull << 20));
    _Float16* qbuf   = (_Float16*)(ws + (16ull << 20));
    _Float16* kbuf   = (_Float16*)(ws + (24ull << 20));
    _Float16* vTbuf  = (_Float16*)(ws + (32ull << 20));
    _Float16* aobuf  = (_Float16*)(ws + (40ull << 20));

    rmsnorm_kernel<<<M_ROWS, 256, 0, stream>>>(x, xn);
    cvt_f32_f16_kernel<<<(3 * C_DIM * C_DIM) / 256, 256, 0, stream>>>(w_qkv, wqkv16, 3 * C_DIM * C_DIM);
    cvt_f32_f16_kernel<<<(C_DIM * C_DIM) / 256, 256, 0, stream>>>(w_out, wout16, C_DIM * C_DIM);
    qkv_gemm_kernel<<<dim3(M_ROWS / 128, (3 * C_DIM) / 64), 256, 0, stream>>>(xn, wqkv16, qbuf, kbuf, vTbuf);
    rope_kernel<<<(BH * T_SEQ * 32) / 256, 256, 0, stream>>>(qbuf, kbuf);
    attn_kernel<<<dim3(BH, T_SEQ / 128), 256, 0, stream>>>(qbuf, kbuf, vTbuf, aobuf);
    out_gemm_kernel<<<dim3(M_ROWS / 128, C_DIM / 64), 256, 0, stream>>>(aobuf, wout16, out);
}
